// DeepSpeedAttention_63556926046676
// MI455X (gfx1250) — matrix operand order repaired and checked by reading, not compiled, not run on hardware
//
#include <hip/hip_runtime.h>
#include <hip/hip_bf16.h>

typedef __attribute__((ext_vector_type(16))) __bf16 v16bf;
typedef __attribute__((ext_vector_type(8)))  float  v8f;

#define NTHREADS 256
#define S_LEN 2048
#define HDIM  2048
#define NHEADS 16
#define DHEAD 128
#define SCALE 0.08838834764831845f     /* 1/sqrt(128) */
#define NEGV  -1.0e9f
#define LOG2E 1.44269504088896340736f

// ---------------------------------------------------------------------------
// fragment helpers
// ---------------------------------------------------------------------------
__device__ __forceinline__ v16bf ld_frag(const __bf16* p0, const __bf16* p1) {
  v16bf r;
  uint4* d = reinterpret_cast<uint4*>(&r);
  d[0] = *reinterpret_cast<const uint4*>(p0);
  d[1] = *reinterpret_cast<const uint4*>(p1);
  return r;
}

__device__ __forceinline__ v8f wmma_bf16(v16bf a, v16bf b, v8f c) {
  // D = A(16x32) * B(32x16) + C, f32 accumulate
  return __builtin_amdgcn_wmma_f32_16x16x32_bf16(false, a, false, b, (short)0, c,
                                                 false, false);
}

union Pack8 { __bf16 h[8]; uint4 u; };

// ---------------------------------------------------------------------------
// fp32 -> bf16 conversion (weights only; x is converted in-GEMM)
// ---------------------------------------------------------------------------
__global__ __launch_bounds__(NTHREADS) void cvt_bf16(const float* __restrict__ in,
                                                     __bf16* __restrict__ out, int n) {
  int i = blockIdx.x * blockDim.x + threadIdx.x;
  if (i < n) out[i] = (__bf16)in[i];
}

// ---------------------------------------------------------------------------
// 128x256 GEMM core, BK=64. A:[M,K] row-major (lda), B:[K,N] row-major (ldb).
// 8 waves laid out 2(M) x 4(N); each wave -> 64x64 = 4x4 WMMA tiles.
// A may be fp32 (converted to bf16 while staging into LDS).
// ---------------------------------------------------------------------------
template <bool A_IS_F32>
__device__ __forceinline__ void gemm_core_128x256(const void* __restrict__ Agv,
                                                  const __bf16* __restrict__ Bg,
                                                  int lda, int ldb, int K,
                                                  v8f (&acc)[4][4]) {
  __shared__ __bf16 As[128][72];    // 64 + 8 pad (144B rows, 16B aligned)
  __shared__ __bf16 Bs[64][264];    // 256 + 8 pad (528B rows, 16B aligned)
  const int t = threadIdx.x;
  const int lane = t & 31, wave = t >> 5;
  const int wm = (wave & 1) * 64;
  const int wn = (wave >> 1) * 64;
  const int half = lane >> 4, ln16 = lane & 15;
  const int arow = t >> 1, acol = (t & 1) * 32;   // 128 rows x 64 cols
  const int brow = t >> 2, bcol = (t & 3) * 64;   // 64 rows x 256 cols

#pragma unroll
  for (int i = 0; i < 4; ++i)
#pragma unroll
    for (int j = 0; j < 4; ++j)
#pragma unroll
      for (int p = 0; p < 8; ++p) acc[i][j][p] = 0.0f;

  for (int k0 = 0; k0 < K; k0 += 64) {
    // ---- stage A tile (convert fp32 -> bf16 on the fly if needed) ----
    if constexpr (A_IS_F32) {
      const float* as = (const float*)Agv + (size_t)arow * lda + (k0 + acol);
#pragma unroll
      for (int g = 0; g < 4; ++g) {
        float4 x0 = reinterpret_cast<const float4*>(as)[g * 2];
        float4 x1 = reinterpret_cast<const float4*>(as)[g * 2 + 1];
        Pack8 pk;
        pk.h[0] = (__bf16)x0.x; pk.h[1] = (__bf16)x0.y;
        pk.h[2] = (__bf16)x0.z; pk.h[3] = (__bf16)x0.w;
        pk.h[4] = (__bf16)x1.x; pk.h[5] = (__bf16)x1.y;
        pk.h[6] = (__bf16)x1.z; pk.h[7] = (__bf16)x1.w;
        *reinterpret_cast<uint4*>(&As[arow][acol + g * 8]) = pk.u;
      }
      if (k0 + 64 < K) __builtin_prefetch(as + 64, 0, 0);
    } else {
      const __bf16* as = (const __bf16*)Agv + (size_t)arow * lda + (k0 + acol);
#pragma unroll
      for (int g = 0; g < 4; ++g)
        *reinterpret_cast<uint4*>(&As[arow][acol + g * 8]) =
            reinterpret_cast<const uint4*>(as)[g];
      if (k0 + 64 < K) __builtin_prefetch(as + 64, 0, 0);
    }
    // ---- stage B tile ----
    const __bf16* bsrc = Bg + (size_t)(k0 + brow) * ldb + bcol;
#pragma unroll
    for (int g = 0; g < 8; ++g)
      *reinterpret_cast<uint4*>(&Bs[brow][bcol + g * 8]) =
          reinterpret_cast<const uint4*>(bsrc)[g];
    if (k0 + 64 < K) __builtin_prefetch(bsrc + (size_t)64 * ldb, 0, 0);
    __syncthreads();

    // ---- 2 x (K=32) WMMA sub-steps ----
#pragma unroll
    for (int ks = 0; ks < 64; ks += 32) {
      v16bf af[4];
#pragma unroll
      for (int i = 0; i < 4; ++i) {
        const __bf16* ap = &As[wm + i * 16 + ln16][ks + half * 8];
        af[i] = ld_frag(ap, ap + 16);
      }
#pragma unroll
      for (int j = 0; j < 4; ++j) {
        v16bf bfj;
#pragma unroll
        for (int s = 0; s < 16; ++s)
          bfj[s] = Bs[ks + half * 8 + (s & 7) + (s >> 3) * 16][wn + j * 16 + ln16];
#pragma unroll
        for (int i = 0; i < 4; ++i) acc[i][j] = wmma_bf16(af[i], bfj, acc[i][j]);
      }
    }
    __syncthreads();
  }
}

// ---------------------------------------------------------------------------
// QKV projection: [4096,2048](f32) x [2048,6144](bf16) + bias -> [3][B,H,S,D] bf16
// ---------------------------------------------------------------------------
__global__ __launch_bounds__(NTHREADS) void qkv_gemm_kernel(const float* __restrict__ X,
                                                            const __bf16* __restrict__ W,
                                                            const float* __restrict__ bias,
                                                            __bf16* __restrict__ qkvh) {
  v8f acc[4][4];
  const int bm = blockIdx.y * 128;
  const int bn = blockIdx.x * 256;
  gemm_core_128x256<true>(X + (size_t)bm * HDIM, W + bn, HDIM, 3 * HDIM, HDIM, acc);

  const int lane = threadIdx.x & 31, wave = threadIdx.x >> 5;
  const int wm = (wave & 1) * 64, wn = (wave >> 1) * 64;
  const int half = lane >> 4, ln16 = lane & 15;
#pragma unroll
  for (int i = 0; i < 4; ++i)
#pragma unroll
    for (int j = 0; j < 4; ++j) {
      const int gn = bn + wn + j * 16 + ln16;
      const float bv = bias[gn];
      const int which = gn >> 11;      // 0:q 1:k 2:v
      const int rem = gn & 2047;
      const int hh = rem >> 7;         // head
      const int dd = rem & 127;        // dim in head
#pragma unroll
      for (int p = 0; p < 8; ++p) {
        const int gm = bm + wm + i * 16 + half * 8 + p;
        const int bb = gm >> 11;       // batch
        const int ss = gm & 2047;      // seq
        const size_t idx = ((((size_t)which * 2 + bb) * NHEADS + hh) * S_LEN + ss) * DHEAD + dd;
        qkvh[idx] = (__bf16)(acc[i][j][p] + bv);
      }
    }
}

// ---------------------------------------------------------------------------
// Flash attention: block = (q-tile of 128, b*h). Wave owns 16 q rows.
// K is transposed into LDS at tile-load time as Ktt[d][key]; its B fragments, like V's,
// are gathered one u16 per row (Q and P use two 16-byte loads).
// ---------------------------------------------------------------------------
__global__ __launch_bounds__(NTHREADS) void flash_attn_kernel(const __bf16* __restrict__ Q,
                                                              const __bf16* __restrict__ Kg,
                                                              const __bf16* __restrict__ Vg,
                                                              __bf16* __restrict__ ctx) {
  const int qt = blockIdx.x;           // 0..15
  const int bh = blockIdx.y;           // 0..31
  const int b = bh >> 4, h = bh & 15;
  const int t = threadIdx.x;
  const int wave = t >> 5, lane = t & 31;
  const int half = lane >> 4, ln16 = lane & 15;

  const size_t head_off = (size_t)bh * (S_LEN * DHEAD);
  const __bf16* qp = Q + head_off;
  const __bf16* kp = Kg + head_off;
  const __bf16* vp = Vg + head_off;

  __shared__ __bf16 Ktt[128][72];      // K transposed: [d][key], 18.4 KB
  __shared__ __bf16 Vt[64][136];       // V row-major:  [key][d], 17.4 KB
  __shared__ __bf16 Pt[8][16][72];     // per-wave P staging, 18.4 KB

  const int qrow0 = qt * 128 + wave * 16;

  // Q fragments for the 4 K-steps over D=128 (registers for the whole kernel)
  v16bf qf[4];
  {
    const __bf16* qrow = qp + (size_t)(qrow0 + ln16) * DHEAD;
#pragma unroll
    for (int ds = 0; ds < 4; ++ds) {
      const __bf16* p0 = qrow + ds * 32 + half * 8;
      qf[ds] = ld_frag(p0, p0 + 16);
    }
  }

  v8f cacc[8];                          // ctx accumulators: 8 d-tiles of 16
#pragma unroll
  for (int i = 0; i < 8; ++i)
#pragma unroll
    for (int p = 0; p < 8; ++p) cacc[i][p] = 0.0f;

  float m_run[8], l_run[8];
#pragma unroll
  for (int p = 0; p < 8; ++p) { m_run[p] = -3.0e38f; l_run[p] = 0.0f; }

  const int r = t >> 2;                 // loader: key row 0..63
  const int c = (t & 3) * 32;           // loader: d col base

  const int ktn = 2 * qt + 2;           // causal: only k-tiles up to the diagonal
  for (int kt = 0; kt < ktn; ++kt) {
    __syncthreads();
    {
      const __bf16* ks = kp + (size_t)(kt * 64 + r) * DHEAD + c;
      const __bf16* vs = vp + (size_t)(kt * 64 + r) * DHEAD + c;
      uint4 kv[4];
      uint4* vd = reinterpret_cast<uint4*>(&Vt[r][c]);
#pragma unroll
      for (int i = 0; i < 4; ++i) {
        kv[i] = reinterpret_cast<const uint4*>(ks)[i];
        vd[i] = reinterpret_cast<const uint4*>(vs)[i];
      }
      // transpose K into LDS: Ktt[d][key]
      const __bf16* ke = reinterpret_cast<const __bf16*>(kv);
#pragma unroll
      for (int i = 0; i < 32; ++i) Ktt[c + i][r] = ke[i];
      if (kt + 1 < ktn) {
        __builtin_prefetch(ks + 64 * DHEAD, 0, 0);
        __builtin_prefetch(vs + 64 * DHEAD, 0, 0);
      }
    }
    __syncthreads();

    // ---- scores: S = Q(16x128) @ K^T(128x64) ----
    v8f sacc[4];
#pragma unroll
    for (int nt = 0; nt < 4; ++nt)
#pragma unroll
      for (int p = 0; p < 8; ++p) sacc[nt][p] = 0.0f;

#pragma unroll
    for (int ds = 0; ds < 4; ++ds) {
#pragma unroll
      for (int nt = 0; nt < 4; ++nt) {
        // B frag: lane = key column (nt*16+ln16), elements = d in table order
        v16bf bfr;
#pragma unroll
        for (int s = 0; s < 16; ++s)
          bfr[s] = Ktt[ds * 32 + half * 8 + (s & 7) + (s >> 3) * 16][nt * 16 + ln16];
        sacc[nt] = wmma_bf16(qf[ds], bfr, sacc[nt]);
      }
    }

    // ---- online softmax (per wave; rows live in 16-lane halves) ----
    float sv[4][8], rmax[8];
#pragma unroll
    for (int p = 0; p < 8; ++p) rmax[p] = -3.0e38f;
#pragma unroll
    for (int nt = 0; nt < 4; ++nt)
#pragma unroll
      for (int p = 0; p < 8; ++p) {
        const int key = kt * 64 + nt * 16 + ln16;
        const int qr = qrow0 + half * 8 + p;
        float v = sacc[nt][p] * SCALE;
        if (key > qr) v = NEGV;
        sv[nt][p] = v;
        rmax[p] = fmaxf(rmax[p], v);
      }
#pragma unroll
    for (int off = 1; off < 16; off <<= 1)
#pragma unroll
      for (int p = 0; p < 8; ++p) rmax[p] = fmaxf(rmax[p], __shfl_xor(rmax[p], off, 32));

    float alpha[8], rsum[8];
#pragma unroll
    for (int p = 0; p < 8; ++p) {
      const float mnew = fmaxf(m_run[p], rmax[p]);
      alpha[p] = exp2f((m_run[p] - mnew) * LOG2E);
      m_run[p] = mnew;
      rsum[p] = 0.0f;
    }
#pragma unroll
    for (int nt = 0; nt < 4; ++nt)
#pragma unroll
      for (int p = 0; p < 8; ++p) {
        const float pe = exp2f((sv[nt][p] - m_run[p]) * LOG2E);
        rsum[p] += pe;
        Pt[wave][half * 8 + p][nt * 16 + ln16] = (__bf16)pe;
      }
#pragma unroll
    for (int off = 1; off < 16; off <<= 1)
#pragma unroll
      for (int p = 0; p < 8; ++p) rsum[p] += __shfl_xor(rsum[p], off, 32);
#pragma unroll
    for (int p = 0; p < 8; ++p) l_run[p] = l_run[p] * alpha[p] + rsum[p];
#pragma unroll
    for (int dt = 0; dt < 8; ++dt)
#pragma unroll
      for (int p = 0; p < 8; ++p) cacc[dt][p] *= alpha[p];

    // wave-local LDS RAW on Pt: DS ops are in-order per wave; force counter wait
    asm volatile("s_wait_dscnt 0" ::: "memory");

    // ---- ctx += P(16x64) @ V(64x128) ----
#pragma unroll
    for (int kk = 0; kk < 2; ++kk) {
      const __bf16* pr = &Pt[wave][ln16][kk * 32 + half * 8];
      v16bf pf = ld_frag(pr, pr + 16);
#pragma unroll
      for (int dt = 0; dt < 8; ++dt) {
        v16bf vfr;   // lane = d column, elements = keys in table order
#pragma unroll
        for (int s = 0; s < 16; ++s)
          vfr[s] = Vt[kk * 32 + half * 8 + (s & 7) + (s >> 3) * 16][dt * 16 + ln16];
        cacc[dt] = wmma_bf16(pf, vfr, cacc[dt]);
      }
    }
  }

  // ---- normalize + store ctx bf16 in [B,S,H] (heads merged) ----
#pragma unroll
  for (int dt = 0; dt < 8; ++dt)
#pragma unroll
    for (int p = 0; p < 8; ++p) {
      const int srow = qrow0 + half * 8 + p;
      const int d = dt * 16 + ln16;
      const float v = cacc[dt][p] / l_run[p];
      ctx[((size_t)b * S_LEN + srow) * HDIM + h * DHEAD + d] = (__bf16)v;
    }
}

// ---------------------------------------------------------------------------
// Output projection: [4096,2048](bf16) x [2048,2048](bf16) + bias -> fp32 out
// ---------------------------------------------------------------------------
__global__ __launch_bounds__(NTHREADS) void out_gemm_kernel(const __bf16* __restrict__ Ctx,
                                                            const __bf16* __restrict__ W,
                                                            const float* __restrict__ bias,
                                                            float* __restrict__ out) {
  v8f acc[4][4];
  const int bm = blockIdx.y * 128;
  const int bn = blockIdx.x * 256;
  gemm_core_128x256<false>(Ctx + (size_t)bm * HDIM, W + bn, HDIM, HDIM, HDIM, acc);

  const int lane = threadIdx.x & 31, wave = threadIdx.x >> 5;
  const int wm = (wave & 1) * 64, wn = (wave >> 1) * 64;
  const int half = lane >> 4, ln16 = lane & 15;
#pragma unroll
  for (int i = 0; i < 4; ++i)
#pragma unroll
    for (int j = 0; j < 4; ++j) {
      const int gn = bn + wn + j * 16 + ln16;
      const float bv = bias[gn];
#pragma unroll
      for (int p = 0; p < 8; ++p) {
        const int gm = bm + wm + i * 16 + half * 8 + p;
        out[(size_t)gm * HDIM + gn] = acc[i][j][p] + bv;
      }
    }
}

// ---------------------------------------------------------------------------
extern "C" void kernel_launch(void* const* d_in, const int* in_sizes, int n_in,
                              void* d_out, int out_size, void* d_ws, size_t ws_size,
                              hipStream_t stream) {
  (void)in_sizes; (void)n_in; (void)out_size; (void)ws_size;
  const float* x    = (const float*)d_in[0];   // [B,S,H]
  const float* wqkv = (const float*)d_in[1];   // [H,3H]
  const float* bqkv = (const float*)d_in[2];   // [3H]
  const float* wo   = (const float*)d_in[3];   // [H,H]
  const float* bo   = (const float*)d_in[4];   // [H]
  float* out = (float*)d_out;

  const size_t nX    = 8388608ull;    // B*S*H
  const size_t nWqkv = 12582912ull;   // H*3H
  const size_t nWo   = 4194304ull;    // H*H

  char* ws = (char*)d_ws;
  __bf16* wqkvb = (__bf16*)ws; ws += nWqkv * 2;
  __bf16* qkvh  = (__bf16*)ws; ws += 3 * nX * 2;   // [3][B,H,S,D]
  __bf16* ctxb  = (__bf16*)ws; ws += nX * 2;
  __bf16* wob   = (__bf16*)ws; ws += nWo * 2;

  cvt_bf16<<<(unsigned)(nWqkv / NTHREADS), NTHREADS, 0, stream>>>(wqkv, wqkvb, (int)nWqkv);
  cvt_bf16<<<(unsigned)(nWo / NTHREADS),   NTHREADS, 0, stream>>>(wo, wob, (int)nWo);

  qkv_gemm_kernel<<<dim3(24, 32), NTHREADS, 0, stream>>>(x, wqkvb, bqkv, qkvh);

  flash_attn_kernel<<<dim3(16, 32), NTHREADS, 0, stream>>>(qkvh, qkvh + nX,
                                                           qkvh + 2 * nX, ctxb);

  out_gemm_kernel<<<dim3(8, 32), NTHREADS, 0, stream>>>(ctxb, wob, bo, out);
}
